// MAMBA_TCM_79706003079173
// MI455X (gfx1250) — matrix operand order repaired and checked by reading, not compiled, not run on hardware
//
#include <hip/hip_runtime.h>
#include <hip/hip_bf16.h>
#include <cmath>

typedef __attribute__((ext_vector_type(16))) _Float16 v16h;
typedef __attribute__((ext_vector_type(8)))  _Float16 v8h;
typedef __attribute__((ext_vector_type(8)))  float    v8f;
typedef __attribute__((ext_vector_type(4)))  int      v4i;
typedef __attribute__((address_space(1)))    v4i      gv4i;   // global int4
typedef __attribute__((address_space(3)))    v4i      lv4i;   // LDS int4

#define B_   32
#define L_   256
#define S_   3000
#define SP_  3008   // S padded to multiple of 32 (WMMA K-step)
#define E_   512
#define DIN_ 1024
#define DST_ 16
#define DTR_ 32
#define NF_  21

// ---- gfx1250 async global->LDS staging (ASYNCcnt-tracked), guarded --------
#if defined(__AMDGCN__) && __has_builtin(__builtin_amdgcn_global_load_async_to_lds_b128)
  #define HAVE_ASYNC_LDS 1
#endif
#if defined(__AMDGCN__) && __has_builtin(__builtin_amdgcn_s_wait_asynccnt)
  #define WAIT_ASYNC0() __builtin_amdgcn_s_wait_asynccnt(0)
#elif defined(HAVE_ASYNC_LDS)
  #define WAIT_ASYNC0() asm volatile("s_wait_asynccnt 0x0" ::: "memory")
#else
  #define WAIT_ASYNC0()
#endif

// ---------------------------------------------------------------------------
// WMMA tiled GEMM: C[M,N] = act(A[M,Kp] @ Bt[N,Kp]^T + bias)
// A: f16 row-major [M][lda], K zero-padded. Bt: f16 [N][ldk] (pre-transposed).
// Block tile 128x64, 8 wave32 waves (4x2), each wave 2x2 of 16x16x32 f16 WMMA.
// Double-buffered LDS; tiles staged with global_load_async_to_lds_b128.
// ACT: 0 none, 1 softplus. C32/C16 written when non-null.
// ---------------------------------------------------------------------------
template<int ACT>
__global__ __launch_bounds__(256)
void gemm_wmma(const _Float16* __restrict__ A, const _Float16* __restrict__ Bt,
               const float* __restrict__ bias,
               float* __restrict__ C32, _Float16* __restrict__ C16,
               int kTiles, int lda, int ldk, int ldc)
{
  constexpr int LDA = 40;   // halves; 80B row stride -> conflict-free a-frags
  constexpr int LDB = 48;   // halves; 96B row stride, 16B-aligned v8h b-frag halves
  __shared__ alignas(64) _Float16 sA[2][128 * LDA];
  __shared__ alignas(64) _Float16 sB[2][64 * LDB];

  const int tid  = threadIdx.x;
  const int lane = tid & 31;
  const int wave = tid >> 5;
  const int wm   = wave >> 1;        // 0..3 : wave M offset 32*wm
  const int wn   = wave & 1;         // 0..1 : wave N offset 32*wn
  const int m0   = blockIdx.y * 128;
  const int n0   = blockIdx.x * 64;

  // stage one 128x32 A tile + 64x32 Bt tile (16B chunks, 3 per thread)
  auto stage = [&](int kt, int buf) {
    const int kbase = kt << 5;
#if defined(HAVE_ASYNC_LDS)
    #pragma unroll
    for (int j = 0; j < 2; ++j) {
      int c = tid + j * 256;                 // 512 chunks of 8 halves
      int m = c >> 2, ko = (c & 3) * 8;
      __builtin_amdgcn_global_load_async_to_lds_b128(
          (gv4i*)(A + (size_t)(m0 + m) * lda + kbase + ko),
          (lv4i*)&sA[buf][m * LDA + ko], 0, 0);
    }
    {
      int n = tid >> 2, ko = (tid & 3) * 8;  // 256 chunks
      __builtin_amdgcn_global_load_async_to_lds_b128(
          (gv4i*)(Bt + (size_t)(n0 + n) * ldk + kbase + ko),
          (lv4i*)&sB[buf][n * LDB + ko], 0, 0);
    }
#else
    #pragma unroll
    for (int j = 0; j < 2; ++j) {
      int c = tid + j * 256;
      int m = c >> 2, ko = (c & 3) * 8;
      *(v8h*)&sA[buf][m * LDA + ko] =
          *(const v8h*)(A + (size_t)(m0 + m) * lda + kbase + ko);
    }
    {
      int n = tid >> 2, ko = (tid & 3) * 8;
      *(v8h*)&sB[buf][n * LDB + ko] =
          *(const v8h*)(Bt + (size_t)(n0 + n) * ldk + kbase + ko);
    }
#endif
  };

  v8f acc[2][2] = {};
  stage(0, 0);

  for (int kt = 0; kt < kTiles; ++kt) {
    const int cur = kt & 1;
    WAIT_ASYNC0();                 // my async copies into 'cur' are done
    __syncthreads();               // everyone's are done; prior reads retired
    if (kt + 1 < kTiles) stage(kt + 1, cur ^ 1);

    const int am = lane & 15;
    const int kh = lane >> 4;      // which K-half this lane holds
    v16h afrag[2], bfrag[2];
    #pragma unroll
    for (int i = 0; i < 2; ++i) {
      const _Float16* ap = &sA[cur][(wm * 32 + i * 16 + am) * LDA];
      v8h lo = *(const v8h*)(ap + kh * 8);        // K = kh*8 ..
      v8h hi = *(const v8h*)(ap + 16 + kh * 8);   // K = 16+kh*8 ..
      afrag[i] = __builtin_shufflevector(lo, hi,
                   0,1,2,3,4,5,6,7,8,9,10,11,12,13,14,15);
    }
    #pragma unroll
    for (int j = 0; j < 2; ++j) {
      const _Float16* bp = &sB[cur][(wn * 32 + j * 16 + am) * LDB];
      v8h lo = *(const v8h*)(bp + kh * 8);
      v8h hi = *(const v8h*)(bp + 16 + kh * 8);
      bfrag[j] = __builtin_shufflevector(lo, hi,
                   0,1,2,3,4,5,6,7,8,9,10,11,12,13,14,15);
    }

    #pragma unroll
    for (int i = 0; i < 2; ++i)
      #pragma unroll
      for (int j = 0; j < 2; ++j)
        acc[i][j] = __builtin_amdgcn_wmma_f32_16x16x32_f16(
            false, afrag[i], false, bfrag[j], (short)0, acc[i][j],
            false, false);
  }

  // epilogue: C/D layout -> row = 8*(lane>>4)+r, col = lane&15
  #pragma unroll
  for (int i = 0; i < 2; ++i) {
    #pragma unroll
    for (int j = 0; j < 2; ++j) {
      #pragma unroll
      for (int r = 0; r < 8; ++r) {
        int row = m0 + wm * 32 + i * 16 + (lane >> 4) * 8 + r;
        int col = n0 + wn * 32 + j * 16 + (lane & 15);
        float v = acc[i][j][r];
        if (bias) v += bias[col];
        if (ACT == 1) v = (v > 20.f) ? v : logf(1.f + expf(v)); // softplus
        if (C32) C32[(size_t)row * ldc + col] = v;
        if (C16) C16[(size_t)row * ldc + col] = (_Float16)v;
      }
    }
  }
}

// ---------------------------------------------------------------------------
// One-time layout conversions (streaming, trivial vs 23.3 TB/s)
// ---------------------------------------------------------------------------
// weights: f32 [K][N] row-major -> f16 [N][Kp] transposed, K zero-padded
__global__ __launch_bounds__(256)
void convert_w_t(const float* __restrict__ src, _Float16* __restrict__ dst,
                 int K, int N, int Kp)
{
  int i = blockIdx.x * 256 + threadIdx.x;
  if (i >= N * Kp) return;
  int n = i / Kp, k = i - n * Kp;
  dst[i] = (_Float16)((k < K) ? src[(size_t)k * N + n] : 0.f);
}
// activations: f32 [R][C] -> f16 [R][Cp], C zero-padded
__global__ __launch_bounds__(256)
void convert_a_pad(const float* __restrict__ src, _Float16* __restrict__ dst,
                   int R, int C, int Cp)
{
  int i = blockIdx.x * 256 + threadIdx.x;
  if (i >= R * Cp) return;
  int r = i / Cp, c = i - r * Cp;
  dst[i] = (_Float16)((c < C) ? src[(size_t)r * C + c] : 0.f);
}

// ---------------------------------------------------------------------------
// BatchNorm over axis 0 (batch) + ELU -> f16 output (feeds next GEMM)
// ---------------------------------------------------------------------------
__global__ __launch_bounds__(256)
void bn_elu(const float* __restrict__ e1, const float* __restrict__ gamma,
            const float* __restrict__ beta, _Float16* __restrict__ out16)
{
  const int i = blockIdx.x * 256 + threadIdx.x;     // i over L*E
  const int e = i & (E_ - 1);
  float s = 0.f, sq = 0.f;
  #pragma unroll 4
  for (int b = 0; b < B_; ++b) {
    float v = e1[(size_t)b * (L_ * E_) + i];
    s += v; sq += v * v;
  }
  const float mu   = s * (1.f / B_);
  const float var  = sq * (1.f / B_) - mu * mu;     // biased var (jnp.var)
  const float rinv = rsqrtf(var + 1e-5f);
  const float g = gamma[e], bt = beta[e];
  for (int b = 0; b < B_; ++b) {
    float v = e1[(size_t)b * (L_ * E_) + i];
    float n = g * (v - mu) * rinv + bt;
    float o = (n > 0.f) ? n : (expf(n) - 1.f);      // ELU
    out16[(size_t)b * (L_ * E_) + i] = (_Float16)o;
  }
}

// ---------------------------------------------------------------------------
// Causal depthwise conv (DCONV=4) + SiLU; emits f32 (scan) and f16 (GEMM A)
// ---------------------------------------------------------------------------
__global__ __launch_bounds__(256)
void conv_silu(const float* __restrict__ xz, const float* __restrict__ Wc,
               const float* __restrict__ bc,
               float* __restrict__ uc, _Float16* __restrict__ uch)
{
  const int i   = blockIdx.x * 256 + threadIdx.x;   // over B*L*DIN
  const int d   = i & (DIN_ - 1);
  const int row = i >> 10;                          // b*L + l
  const int l   = row & (L_ - 1);
  float acc = bc[d];
  #pragma unroll
  for (int k = 0; k < 4; ++k) {
    int ls = l - 3 + k;
    if (ls >= 0) acc += xz[(size_t)(row - 3 + k) * (2 * DIN_) + d] * Wc[k * DIN_ + d];
  }
  float s = acc / (1.f + expf(-acc));               // silu
  uc[i]  = s;
  uch[i] = (_Float16)s;
}

// ---------------------------------------------------------------------------
// Selective scan. One thread per (b,d), 16 states in registers, B_t/C_t in
// LDS. Only the last step's y is needed -> fuse +uc*D and *silu(z) there.
// ---------------------------------------------------------------------------
__global__ __launch_bounds__(256)
void mamba_scan(const float* __restrict__ dtbuf, const float* __restrict__ xdb,
                const float* __restrict__ ucbuf, const float* __restrict__ xz,
                const float* __restrict__ A_log, const float* __restrict__ Dp,
                float* __restrict__ ylast)
{
  const int b = blockIdx.y;
  const int d = blockIdx.x * 256 + threadIdx.x;
  __shared__ float sBC[32];

  float Arow[DST_], h[DST_];
  #pragma unroll
  for (int s = 0; s < DST_; ++s) {
    Arow[s] = -expf(A_log[d * DST_ + s]);           // A = -exp(A_log)
    h[s] = 0.f;
  }
  const float Dval = Dp[d];

  for (int l = 0; l < L_; ++l) {
    const int row = b * L_ + l;
    __syncthreads();
    if (threadIdx.x < 32) sBC[threadIdx.x] = xdb[(size_t)row * 64 + 32 + threadIdx.x];
    __syncthreads();
    const float dt = dtbuf[(size_t)row * DIN_ + d];
    const float u  = ucbuf[(size_t)row * DIN_ + d];
    const float du = dt * u;
    float y = 0.f;
    #pragma unroll
    for (int s = 0; s < DST_; ++s) {
      float dA = expf(dt * Arow[s]);
      h[s] = dA * h[s] + du * sBC[s];
      y += h[s] * sBC[16 + s];
    }
    if (l == L_ - 1) {
      float z  = xz[(size_t)row * (2 * DIN_) + DIN_ + d];
      float sz = z / (1.f + expf(-z));
      ylast[b * DIN_ + d] = (y + u * Dval) * sz;
    }
  }
}

// ---------------------------------------------------------------------------
// Head: out_proj (last step only) -> proj -> concat eeg -> classifier.
// ---------------------------------------------------------------------------
__global__ __launch_bounds__(256)
void head(const float* __restrict__ ylast, const float* __restrict__ Wout,
          const float* __restrict__ Wproj, const float* __restrict__ bproj,
          const float* __restrict__ eeg,   const float* __restrict__ Wcls,
          const float* __restrict__ bcls,  float* __restrict__ out)
{
  const int b = blockIdx.x;
  const int t = threadIdx.x;
  __shared__ float sy[DIN_];
  __shared__ float sm[E_];
  __shared__ float sp[51];
  #pragma unroll
  for (int j = 0; j < 4; ++j) sy[t + j * 256] = ylast[b * DIN_ + t + j * 256];
  __syncthreads();
  #pragma unroll
  for (int j = 0; j < 2; ++j) {
    const int c = t + j * 256;
    float acc = 0.f;
    for (int d = 0; d < DIN_; ++d) acc += sy[d] * Wout[(size_t)d * E_ + c];
    sm[c] = acc;
  }
  __syncthreads();
  if (t < 30) {
    float acc = bproj[t];
    for (int e = 0; e < E_; ++e) acc += sm[e] * Wproj[e * 30 + t];
    sp[t] = acc;
  } else if (t < 51) {
    sp[t] = eeg[b * NF_ + (t - 30)];
  }
  __syncthreads();
  if (t < 5) {
    float acc = bcls[t];
    for (int i = 0; i < 51; ++i) acc += sp[i] * Wcls[i * 5 + t];
    out[((size_t)b * L_ + (L_ - 1)) * 5 + t] = acc;
  }
}

__global__ void zero_out(float* __restrict__ p, int n)
{
  int i = blockIdx.x * 256 + threadIdx.x;
  if (i < n) p[i] = 0.f;
}

// ---------------------------------------------------------------------------
extern "C" void kernel_launch(void* const* d_in, const int* in_sizes, int n_in,
                              void* d_out, int out_size, void* d_ws, size_t ws_size,
                              hipStream_t stream)
{
  (void)in_sizes; (void)n_in; (void)out_size; (void)ws_size;

  const float* x          = (const float*)d_in[0];
  const float* eeg        = (const float*)d_in[1];
  const float* backbone_W = (const float*)d_in[2];
  const float* backbone_b = (const float*)d_in[3];
  const float* emb_W1     = (const float*)d_in[4];
  const float* emb_b1     = (const float*)d_in[5];
  const float* bn_gamma   = (const float*)d_in[6];
  const float* bn_beta    = (const float*)d_in[7];
  const float* emb_W2     = (const float*)d_in[8];
  const float* emb_b2     = (const float*)d_in[9];
  const float* in_proj_W  = (const float*)d_in[10];
  const float* conv_W     = (const float*)d_in[11];
  const float* conv_b     = (const float*)d_in[12];
  const float* x_proj_W   = (const float*)d_in[13];
  const float* dt_proj_W  = (const float*)d_in[14];
  const float* dt_proj_b  = (const float*)d_in[15];
  const float* A_log      = (const float*)d_in[16];
  const float* Dp         = (const float*)d_in[17];
  const float* out_proj_W = (const float*)d_in[18];
  const float* proj_W     = (const float*)d_in[19];
  const float* proj_b     = (const float*)d_in[20];
  const float* cls_W      = (const float*)d_in[21];
  const float* cls_b      = (const float*)d_in[22];
  float* out = (float*)d_out;

  const int M = B_ * L_;                 // 8192
  char* base = (char*)d_ws;
  size_t off = 0;
  auto alloc = [&](size_t bytes) {
    void* p = base + off;
    off = (off + bytes + 255) & ~(size_t)255;
    return p;
  };
  // f16 operand buffers
  _Float16* xh   = (_Float16*)alloc((size_t)M * SP_ * 2);     // x, K-padded
  _Float16* bwh  = (_Float16*)alloc((size_t)E_ * SP_ * 2);    // backbone_W^T
  _Float16* w1t  = (_Float16*)alloc((size_t)E_ * E_ * 2);
  _Float16* w2t  = (_Float16*)alloc((size_t)E_ * E_ * 2);
  _Float16* inpt = (_Float16*)alloc((size_t)(2 * DIN_) * E_ * 2);
  _Float16* xpt  = (_Float16*)alloc((size_t)64 * DIN_ * 2);
  _Float16* dtpt = (_Float16*)alloc((size_t)DIN_ * DTR_ * 2);
  _Float16* h16  = (_Float16*)alloc((size_t)M * E_ * 2);
  _Float16* e2h  = (_Float16*)alloc((size_t)M * E_ * 2);
  _Float16* eh   = (_Float16*)alloc((size_t)M * E_ * 2);
  _Float16* uch  = (_Float16*)alloc((size_t)M * DIN_ * 2);
  _Float16* xdbh = (_Float16*)alloc((size_t)M * 64 * 2);
  // f32 buffers
  float* e1  = (float*)alloc((size_t)M * E_ * 4);
  float* xz  = (float*)alloc((size_t)M * 2 * DIN_ * 4);
  float* uc  = (float*)alloc((size_t)M * DIN_ * 4);
  float* xdb = (float*)alloc((size_t)M * 64 * 4);
  float* dt  = (float*)alloc((size_t)M * DIN_ * 4);
  float* yl  = (float*)alloc((size_t)B_ * DIN_ * 4);

  const dim3 blk(256);
  auto grid1 = [](int n) { return dim3((n + 255) / 256); };

  zero_out<<<grid1(B_ * L_ * 5), blk, 0, stream>>>(out, B_ * L_ * 5);

  // one-time layout conversions
  convert_a_pad<<<grid1(M * SP_), blk, 0, stream>>>(x, xh, M, S_, SP_);
  convert_w_t<<<grid1(E_ * SP_), blk, 0, stream>>>(backbone_W, bwh, S_, E_, SP_);
  convert_w_t<<<grid1(E_ * E_), blk, 0, stream>>>(emb_W1, w1t, E_, E_, E_);
  convert_w_t<<<grid1(E_ * E_), blk, 0, stream>>>(emb_W2, w2t, E_, E_, E_);
  convert_w_t<<<grid1(2 * DIN_ * E_), blk, 0, stream>>>(in_proj_W, inpt, E_, 2 * DIN_, E_);
  convert_w_t<<<grid1(64 * DIN_), blk, 0, stream>>>(x_proj_W, xpt, DIN_, 64, DIN_);
  convert_w_t<<<grid1(DIN_ * DTR_), blk, 0, stream>>>(dt_proj_W, dtpt, DTR_, DIN_, DTR_);

  // h16 = x @ backbone_W + b            (8192 x 3008) x (3008 x 512), f16 out
  gemm_wmma<0><<<dim3(E_ / 64, M / 128), blk, 0, stream>>>(
      xh, bwh, backbone_b, nullptr, h16, SP_ / 32, SP_, SP_, E_);
  // e1 = h @ emb_W1 + b1  (f32 out for BN)
  gemm_wmma<0><<<dim3(E_ / 64, M / 128), blk, 0, stream>>>(
      h16, w1t, emb_b1, e1, nullptr, E_ / 32, E_, E_, E_);
  // e2 = elu(bn(e1)) -> f16
  bn_elu<<<grid1(L_ * E_), blk, 0, stream>>>(e1, bn_gamma, bn_beta, e2h);
  // e = e2 @ emb_W2 + b2 -> f16
  gemm_wmma<0><<<dim3(E_ / 64, M / 128), blk, 0, stream>>>(
      e2h, w2t, emb_b2, nullptr, eh, E_ / 32, E_, E_, E_);
  // xz = e @ in_proj_W -> f32           (8192 x 512) x (512 x 2048)
  gemm_wmma<0><<<dim3((2 * DIN_) / 64, M / 128), blk, 0, stream>>>(
      eh, inpt, nullptr, xz, nullptr, E_ / 32, E_, E_, 2 * DIN_);
  // uc = silu(causal_conv(u)) -> f32 + f16
  conv_silu<<<grid1(M * DIN_), blk, 0, stream>>>(xz, conv_W, conv_b, uc, uch);
  // xdb = uc @ x_proj_W -> f32 + f16    (8192 x 1024) x (1024 x 64)
  gemm_wmma<0><<<dim3(1, M / 128), blk, 0, stream>>>(
      uch, xpt, nullptr, xdb, xdbh, DIN_ / 32, DIN_, DIN_, 64);
  // dt = softplus(xdb[:, :32] @ dt_proj_W + b)   (K=32, lda=64)
  gemm_wmma<1><<<dim3(DIN_ / 64, M / 128), blk, 0, stream>>>(
      xdbh, dtpt, dt_proj_b, dt, nullptr, 1, 64, DTR_, DIN_);
  // selective scan -> ylast
  mamba_scan<<<dim3(DIN_ / 256, B_), blk, 0, stream>>>(
      dt, xdb, uc, xz, A_log, Dp, yl);
  // out_proj + proj + concat(eeg) + classifier at l = L-1
  head<<<B_, blk, 0, stream>>>(yl, out_proj_W, proj_W, proj_b, eeg,
                               cls_W, cls_b, out);
}
